// GraphConvolution_att_6339371729613
// MI455X (gfx1250) — matrix operand order repaired and checked by reading, not compiled, not run on hardware
//
#include <hip/hip_runtime.h>
#include <hip/hip_bf16.h>
#include <hip/hip_fp16.h>

typedef __attribute__((ext_vector_type(16))) _Float16 v16h;
typedef __attribute__((ext_vector_type(8)))  _Float16 v8h;
typedef __attribute__((ext_vector_type(8)))  float    v8f;
typedef __attribute__((ext_vector_type(4)))  int      v4i;

#define ALPHA_LR 0.2f
static constexpr int B_ = 8, N_ = 2048, F_ = 128;

// Async global->LDS path (gfx1250): guarded so the file always compiles.
#if defined(__gfx1250__) &&                                               \
    __has_builtin(__builtin_amdgcn_global_load_async_to_lds_b128) &&      \
    __has_builtin(__builtin_amdgcn_s_wait_asynccnt)
#define USE_ASYNC_LDS 1
typedef __attribute__((address_space(1))) v4i glb_v4i;   // global (AS1) v4i
typedef __attribute__((address_space(3))) v4i lds_v4i;   // LDS (AS3) v4i
#endif

// ---------------------------------------------------------------------------
// Kernel 1: Wh = x @ W   (f16 inputs, f32 WMMA accumulate, f16 output)
// Block = 128 threads (4 waves), each wave owns a 16-row tile (64 rows/block).
// W is transposed into LDS as f16 so B fragments are contiguous-K loads.
// ---------------------------------------------------------------------------
#define WT_STRIDE 136   // halfs per LDS row; 272 B (16B-aligned, conflict-free)

__global__ __launch_bounds__(128) void k_gemm_xw(const float* __restrict__ x,
                                                 const float* __restrict__ W,
                                                 _Float16* __restrict__ Wh)
{
    __shared__ __align__(16) _Float16 WT[128 * WT_STRIDE]; // WT[e][k]
    const int t = threadIdx.x;

    // cooperative transpose: thread t owns row k=t of W (128 f32, contiguous)
    for (int e0 = 0; e0 < 128; e0 += 4) {
        float4 w = *(const float4*)(W + t * 128 + e0);
        WT[(e0 + 0) * WT_STRIDE + t] = (_Float16)w.x;
        WT[(e0 + 1) * WT_STRIDE + t] = (_Float16)w.y;
        WT[(e0 + 2) * WT_STRIDE + t] = (_Float16)w.z;
        WT[(e0 + 3) * WT_STRIDE + t] = (_Float16)w.w;
    }
    __syncthreads();

    const int wave = t >> 5;
    const int lane = t & 31;
    const int half = lane >> 4;
    const int lrow = lane & 15;
    const long rowBase = (long)blockIdx.x * 64 + wave * 16; // global row base
    const long row = rowBase + lrow;

    v8f acc[8] = {};

    for (int kc = 0; kc < 4; ++kc) {
        // A fragment 16x32 f16: lane row = lrow; K = e + 8*(e>=8) + 8*half
        const float* xr = x + row * F_ + kc * 32 + 8 * half;
        float4 r0 = *(const float4*)(xr + 0);
        float4 r1 = *(const float4*)(xr + 4);
        float4 r2 = *(const float4*)(xr + 16);
        float4 r3 = *(const float4*)(xr + 20);
        v16h a;
        a[0]  = (_Float16)r0.x; a[1]  = (_Float16)r0.y;
        a[2]  = (_Float16)r0.z; a[3]  = (_Float16)r0.w;
        a[4]  = (_Float16)r1.x; a[5]  = (_Float16)r1.y;
        a[6]  = (_Float16)r1.z; a[7]  = (_Float16)r1.w;
        a[8]  = (_Float16)r2.x; a[9]  = (_Float16)r2.y;
        a[10] = (_Float16)r2.z; a[11] = (_Float16)r2.w;
        a[12] = (_Float16)r3.x; a[13] = (_Float16)r3.y;
        a[14] = (_Float16)r3.z; a[15] = (_Float16)r3.w;

#pragma unroll
        for (int tl = 0; tl < 8; ++tl) {
            // B fragment 32x16: lane col = lrow; K = e + 8*(e>=8) + 8*half
            const _Float16* bp = &WT[(tl * 16 + lrow) * WT_STRIDE + kc * 32 + 8 * half];
            v8h lo = *(const v8h*)(bp);
            v8h hi = *(const v8h*)(bp + 16);
            v16h b;
#pragma unroll
            for (int i = 0; i < 8; ++i) { b[i] = lo[i]; b[8 + i] = hi[i]; }
            acc[tl] = __builtin_amdgcn_wmma_f32_16x16x32_f16(false, a, false, b,
                                                             (short)0, acc[tl],
                                                             false, false);
        }
    }

    // store Wh row-major f16: element g -> row g + 8*half, col = tl*16 + lrow
#pragma unroll
    for (int tl = 0; tl < 8; ++tl) {
#pragma unroll
        for (int g = 0; g < 8; ++g) {
            long r = rowBase + g + 8 * half;
            Wh[r * F_ + tl * 16 + lrow] = (_Float16)acc[tl][g];
        }
    }
}

// ---------------------------------------------------------------------------
// Kernel 2: s1 = Wh*a1, s2 = Wh*a2; also write transposed WhT[b][f][n] (f16)
// ---------------------------------------------------------------------------
__global__ __launch_bounds__(256) void k_scores(const _Float16* __restrict__ Wh,
                                                const float* __restrict__ a,
                                                _Float16* __restrict__ WhT,
                                                float* __restrict__ s1,
                                                float* __restrict__ s2)
{
    long r = (long)blockIdx.x * blockDim.x + threadIdx.x; // 0..B*N-1
    const int b = (int)(r >> 11);
    const int n = (int)(r & (N_ - 1));
    const _Float16* wr = Wh + r * F_;
    _Float16* wt = WhT + (long)b * F_ * N_ + n;
    float acc1 = 0.f, acc2 = 0.f;
#pragma unroll 8
    for (int e = 0; e < F_; ++e) {
        _Float16 wh = wr[e];
        float w = (float)wh;
        acc1 += w * a[e];
        acc2 += w * a[F_ + e];
        wt[(long)e * N_] = wh;
    }
    s1[r] = acc1;
    s2[r] = acc2;
}

// ---------------------------------------------------------------------------
// Kernel 2b: per-batch max of s2 (upper bound for overflow-free exp)
// ---------------------------------------------------------------------------
__global__ __launch_bounds__(256) void k_maxs2(const float* __restrict__ s2,
                                               float* __restrict__ M2)
{
    __shared__ float red[256];
    const int b = blockIdx.x, t = threadIdx.x;
    float m = -3.4e38f;
    for (int i = t; i < N_; i += 256) m = fmaxf(m, s2[b * N_ + i]);
    red[t] = m;
    __syncthreads();
    for (int s = 128; s > 0; s >>= 1) {
        if (t < s) red[t] = fmaxf(red[t], red[t + s]);
        __syncthreads();
    }
    if (t == 0) M2[b] = red[0];
}

// ---------------------------------------------------------------------------
// Kernel 3: fused masked-softmax attention:  out = elu((P@Wh)/rowsum) + x
// P = adj ? exp(lrelu(s1+s2) - shift) : 0   with shift = lrelu(s1 + max s2).
// Block = 128 threads (4 waves); wave owns 16 rows; loop over columns in 32s.
// adj is streamed with double-buffered GLOBAL_LOAD_ASYNC_TO_LDS_B128 when
// available (ASYNCcnt-tracked), else with non-temporal direct loads.
// ---------------------------------------------------------------------------
#define ADJ_ROW_I 36                       // ints per LDS adj row (144 B, pad)

__global__ __launch_bounds__(128) void k_attn(const int* __restrict__ adj,
                                              const _Float16* __restrict__ WhT,
                                              const float* __restrict__ s1,
                                              const float* __restrict__ s2,
                                              const float* __restrict__ M2,
                                              const float* __restrict__ x,
                                              float* __restrict__ out)
{
    const int b    = blockIdx.x >> 5;   // 32 row-tiles of 64 per batch
    const int tile = blockIdx.x & 31;
    const int t    = threadIdx.x;
    const int wave = t >> 5, lane = t & 31, half = lane >> 4, lrow = lane & 15;

    const int rowL = tile * 64 + wave * 16 + lrow;   // batch-local row of lane
    const float s1r = s1[b * N_ + rowL];
    const float sb  = s1r + M2[b];
    const float shift = fmaxf(sb, ALPHA_LR * sb);    // lrelu(s1 + max s2)
    const float* s2b = s2 + b * N_;
    const _Float16* WhTb = WhT + (long)b * F_ * N_;

    v8f acc[8] = {};
    float rsum = 0.f;

#if defined(USE_ASYNC_LDS)
    // double-buffered adj staging: [buf][wave][row][36 ints]
    __shared__ __align__(16) int adjLds[2 * 4 * 16 * ADJ_ROW_I];
    const int* adjW = adj + ((long)b * N_ + tile * 64 + wave * 16) * (long)N_;

    // producer: lane covers 4 x b128 chunks; chunk c = i*32+lane ->
    // row = c/8, 16B-column = c%8 (rows fully coalesced: 4 rows x 128B per i)
    const int pr[4] = { (0 * 32 + lane) >> 3, (1 * 32 + lane) >> 3,
                        (2 * 32 + lane) >> 3, (3 * 32 + lane) >> 3 };
    const int pc = (lane & 7) * 4;          // int column within 32-col chunk

#define ISSUE_ADJ(bufi, mcol)                                                  \
    _Pragma("unroll")                                                          \
    for (int i = 0; i < 4; ++i) {                                              \
        const int* srcp = adjW + (long)pr[i] * N_ + (mcol) + pc;               \
        int* dstp = &adjLds[(((bufi) * 4 + wave) * 16 + pr[i]) * ADJ_ROW_I + pc]; \
        __builtin_amdgcn_global_load_async_to_lds_b128((glb_v4i*)srcp,         \
                                                       (lds_v4i*)dstp, 0, 0);  \
    }

    ISSUE_ADJ(0, 0)
#else
    const int* adjR = adj + ((long)b * N_ + rowL) * (long)N_;
#endif

    for (int m = 0; m < N_; m += 32) {
        int av[16];
#if defined(USE_ASYNC_LDS)
        const int cur = (m >> 5) & 1;
        if (m + 32 < N_) {
            ISSUE_ADJ(cur ^ 1, m + 32)
            __builtin_amdgcn_s_wait_asynccnt(4);   // previous tile complete
        } else {
            __builtin_amdgcn_s_wait_asynccnt(0);
        }
        asm volatile("" ::: "memory");
        const int rbase = ((cur * 4 + wave) * 16 + lrow) * ADJ_ROW_I;
        v4i a0 = *(const v4i*)&adjLds[rbase + 8 * half];
        v4i a1 = *(const v4i*)&adjLds[rbase + 8 * half + 4];
        v4i a2 = *(const v4i*)&adjLds[rbase + 8 * half + 16];
        v4i a3 = *(const v4i*)&adjLds[rbase + 8 * half + 20];
#else
        const int* ap = adjR + m + 8 * half;
        v4i a0 = __builtin_nontemporal_load((const v4i*)(ap + 0));
        v4i a1 = __builtin_nontemporal_load((const v4i*)(ap + 4));
        v4i a2 = __builtin_nontemporal_load((const v4i*)(ap + 16));
        v4i a3 = __builtin_nontemporal_load((const v4i*)(ap + 20));
        if (m + 32 < N_) __builtin_prefetch(ap + 32, 0, 1);
#endif
#pragma unroll
        for (int i = 0; i < 4; ++i) {
            av[i] = a0[i]; av[4 + i] = a1[i]; av[8 + i] = a2[i]; av[12 + i] = a3[i];
        }

        v16h A;
#pragma unroll
        for (int e = 0; e < 16; ++e) {
            const int K = e + 8 * (e >= 8) + 8 * half;  // A-fragment K mapping
            float lg = s1r + s2b[m + K];
            float el = fmaxf(lg, ALPHA_LR * lg);        // leaky_relu
            float p  = (av[e] > 0) ? __expf(el - shift) : 0.f;
            _Float16 ph = (_Float16)p;
            rsum += (float)ph;                          // sum exactly what WMMA sees
            A[e] = ph;
        }

#pragma unroll
        for (int tl = 0; tl < 8; ++tl) {
            // B fragment: feature col = tl*16 + lrow, K = e + 8*(e>=8) + 8*half
            const _Float16* bp = WhTb + (long)(tl * 16 + lrow) * N_ + m + 8 * half;
            v8h lo = *(const v8h*)(bp);
            v8h hi = *(const v8h*)(bp + 16);
            v16h Bv;
#pragma unroll
            for (int i = 0; i < 8; ++i) { Bv[i] = lo[i]; Bv[8 + i] = hi[i]; }
            acc[tl] = __builtin_amdgcn_wmma_f32_16x16x32_f16(false, A, false, Bv,
                                                             (short)0, acc[tl],
                                                             false, false);
        }
    }

    // combine the two half-lane partial sums of each row, then broadcast
    rsum += __shfl_xor(rsum, 16, 32);
    float rs[8];
#pragma unroll
    for (int g = 0; g < 8; ++g) rs[g] = __shfl(rsum, g + 8 * half, 32);

    const float* xb = x + (long)b * N_ * F_;
    float* ob = out + (long)b * N_ * F_;
#pragma unroll
    for (int tl = 0; tl < 8; ++tl) {
#pragma unroll
        for (int g = 0; g < 8; ++g) {
            const int r = tile * 64 + wave * 16 + g + 8 * half;
            float h = acc[tl][g] / rs[g];
            float y = h > 0.f ? h : (__expf(h) - 1.f);  // elu
            const long idx = (long)r * F_ + tl * 16 + lrow;
            ob[idx] = y + xb[idx];
        }
    }
}

// ---------------------------------------------------------------------------
extern "C" void kernel_launch(void* const* d_in, const int* in_sizes, int n_in,
                              void* d_out, int out_size, void* d_ws, size_t ws_size,
                              hipStream_t stream)
{
    (void)in_sizes; (void)n_in; (void)out_size; (void)ws_size;
    const float* x   = (const float*)d_in[0];
    const int*   adj = (const int*)d_in[1];
    const float* W   = (const float*)d_in[2];
    const float* a   = (const float*)d_in[3];
    float* out = (float*)d_out;

    char* ws = (char*)d_ws;
    const size_t whBytes = (size_t)B_ * N_ * F_ * sizeof(_Float16); // 4 MB
    _Float16* Wh  = (_Float16*)(ws);
    _Float16* WhT = (_Float16*)(ws + whBytes);
    float* s1 = (float*)(ws + 2 * whBytes);
    float* s2 = s1 + (size_t)B_ * N_;
    float* M2 = s2 + (size_t)B_ * N_;

    k_gemm_xw<<<(B_ * N_) / 64, 128, 0, stream>>>(x, W, Wh);
    k_scores<<<(B_ * N_) / 256, 256, 0, stream>>>(Wh, a, WhT, s1, s2);
    k_maxs2<<<B_, 256, 0, stream>>>(s2, M2);
    k_attn<<<B_ * (N_ / 64), 128, 0, stream>>>(adj, WhT, s1, s2, M2, x, out);
}
